// DiversityMemory_42958262894874
// MI455X (gfx1250) — matrix operand order repaired and checked by reading, not compiled, not run on hardware
//
#include <hip/hip_runtime.h>

// C[256, 65536] = inputs[256,2048] @ features[65536,2048]^T  (f32 in/out)
// Strategy: bf16x3 split-precision WMMA (Ahi*Bhi + Alo*Bhi + Ahi*Blo), f32 accum.
// features streamed once from HBM (512MB dominates => ~25us roofline @ 23.3TB/s).

typedef __attribute__((ext_vector_type(16))) __bf16 v16bf;
typedef __attribute__((ext_vector_type(8)))  __bf16 v8bf;
typedef __attribute__((ext_vector_type(8)))  float  v8f;
typedef __attribute__((ext_vector_type(4)))  float  v4f;

#define BQ 256
#define DQ 2048
#define NQ 65536

// Kernel 1: split f32 inputs into bf16 hi/lo planes (x ~= hi + lo), row-major.
__global__ void cm_split_inputs(const float* __restrict__ in,
                                __bf16* __restrict__ hi,
                                __bf16* __restrict__ lo, int n) {
  int i = blockIdx.x * blockDim.x + threadIdx.x;
  if (i < n) {
    float x = in[i];
    __bf16 h = (__bf16)x;
    hi[i] = h;
    lo[i] = (__bf16)(x - (float)h);
  }
}

// Kernel 2: WMMA GEMM.
// Block = 8 waves: wave = (mgroup in 0..1) * 4 + (ngroup in 0..3).
// Each wave: 8 M-tiles (rows mgroup*128 + t*16 .. +15), one 16-wide N slice.
// Block covers M=256, N=64; grid.x = 65536/64 = 1024.
__global__ __launch_bounds__(256)
void cm_gemm_wmma(const __bf16* __restrict__ Ahi,
                  const __bf16* __restrict__ Alo,
                  const float*  __restrict__ Feat,
                  float* __restrict__ Out) {
  const int lane   = threadIdx.x & 31;
  const int wave   = threadIdx.x >> 5;
  const int mgroup = wave >> 2;      // 0..1
  const int ngroup = wave & 3;       // 0..3
  const int l15    = lane & 15;
  const int half   = lane >> 4;      // 0: lanes 0-15, 1: lanes 16-31

  const int n0    = blockIdx.x * 64 + ngroup * 16;
  const int mbase = mgroup * 128;

  v8f acc[8] = {};

  // B fragment (32x16, K-major per lane): lane l15 holds column N=l15;
  // lanes 0-15 hold K=k0+0..15, lanes 16-31 hold K=k0+16..31 -> one contiguous
  // 64B f32 load from features row (n0+l15).
  const float* featRow = Feat + (size_t)(n0 + l15) * DQ + half * 16;

  // A fragment (16x32 bf16): lanes 0-15 row M=l15 hold K 0..15 (v16bf[0..15]);
  // lanes 16-31 hold K 16..31.
  const size_t aoff = (size_t)(mbase + l15) * DQ + half * 16;

  for (int k0 = 0; k0 < DQ; k0 += 32) {
    // ---- load + split-convert B fragment (reused by all 8 M-tiles) ----
    float xb[16];
    {
      const v4f* p = (const v4f*)(featRow + k0);
      v4f q0 = p[0], q1 = p[1], q2 = p[2], q3 = p[3];
#pragma unroll
      for (int i = 0; i < 4; ++i) {
        xb[i] = q0[i]; xb[4 + i] = q1[i]; xb[8 + i] = q2[i]; xb[12 + i] = q3[i];
      }
    }
    v16bf bhi, blo;
#pragma unroll
    for (int i = 0; i < 16; ++i) {
      __bf16 h = (__bf16)xb[i];
      bhi[i] = h;
      blo[i] = (__bf16)(xb[i] - (float)h);
    }

    // ---- 8 M-tiles, 3 WMMA terms each ----
#pragma unroll
    for (int t = 0; t < 8; ++t) {
      const __bf16* ah = Ahi + aoff + (size_t)t * 16 * DQ + k0;
      const __bf16* al = Alo + aoff + (size_t)t * 16 * DQ + k0;
      v8bf ah0 = *(const v8bf*)ah;
      v8bf ah1 = *(const v8bf*)(ah + 8);
      v8bf al0 = *(const v8bf*)al;
      v8bf al1 = *(const v8bf*)(al + 8);
      v16bf fa_hi, fa_lo;
#pragma unroll
      for (int i = 0; i < 8; ++i) {
        fa_hi[i] = ah0[i]; fa_hi[i + 8] = ah1[i];
        fa_lo[i] = al0[i]; fa_lo[i + 8] = al1[i];
      }
      acc[t] = __builtin_amdgcn_wmma_f32_16x16x32_bf16(
          false, fa_hi, false, bhi, (short)0, acc[t], false, false);
      acc[t] = __builtin_amdgcn_wmma_f32_16x16x32_bf16(
          false, fa_lo, false, bhi, (short)0, acc[t], false, false);
      acc[t] = __builtin_amdgcn_wmma_f32_16x16x32_bf16(
          false, fa_hi, false, blo, (short)0, acc[t], false, false);
    }
  }

  // ---- store C: lane<16 holds rows m..m+7, lane>=16 rows m+8..m+15, N=l15 ----
#pragma unroll
  for (int t = 0; t < 8; ++t) {
    const int mrow = mbase + t * 16 + half * 8;
#pragma unroll
    for (int r = 0; r < 8; ++r) {
      Out[(size_t)(mrow + r) * NQ + (n0 + l15)] = acc[t][r];
    }
  }
}

extern "C" void kernel_launch(void* const* d_in, const int* in_sizes, int n_in,
                              void* d_out, int out_size, void* d_ws, size_t ws_size,
                              hipStream_t stream) {
  // setup_inputs order: inputs, inputs_ema, indexes, features
  const float* inputs = (const float*)d_in[0];
  const float* feats  = (const float*)d_in[3];
  float* out = (float*)d_out;

  __bf16* Ahi = (__bf16*)d_ws;                 // 256*2048 bf16 = 1 MB
  __bf16* Alo = Ahi + (size_t)BQ * DQ;         // +1 MB

  const int n = BQ * DQ;
  cm_split_inputs<<<(n + 255) / 256, 256, 0, stream>>>(inputs, Ahi, Alo, n);
  cm_gemm_wmma<<<NQ / 64, 256, 0, stream>>>(Ahi, Alo, feats, out);
}
